// VectorQuantizer_36988258353665
// MI455X (gfx1250) — matrix operand order repaired and checked by reading, not compiled, not run on hardware
//
#include <hip/hip_runtime.h>

// ---------------------------------------------------------------------------
// VQ forward for MI455X (gfx1250): bf16 WMMA GEMMs + fused argmax search.
// Round 3: vq_sim software-pipelined with two half-pass B buffers so loads
// for one half overlap WMMAs of the other (incl. across pass boundaries).
// vq_sim placed first in the file so the disasm snippet shows the hot loop.
// ---------------------------------------------------------------------------

typedef __attribute__((ext_vector_type(16))) __bf16 v16bf;
typedef __attribute__((ext_vector_type(8)))  float  v8f;
typedef __attribute__((ext_vector_type(4)))  float  f32x4;
typedef __attribute__((ext_vector_type(4)))  unsigned int u32x4;
typedef unsigned short u16;
typedef unsigned int   u32;

#define N_   4
#define C_   512
#define L_   4096
#define D_   256
#define K_   8192
#define NL_  16384          // N_*L_
#define Y_SIZE_ 8388608     // N_*C_*L_
#define EPS_ 1e-8f

// workspace layout (byte offsets)
#define WS_WIN   0u          // D_*C_ u16  (256 KB)
#define WS_WOUT  262144u     // C_*D_ u16  (256 KB)
#define WS_CBN   524288u     // K_*D_ u16  (4 MB)   normalized codebook bf16
#define WS_ZE    4718592u    // NL_*D_ f32 (16 MB)  z_e row-major (p, d)
#define WS_ZEN   21495808u   // NL_*D_ u16 (8 MB)   normalized z_e bf16
#define WS_IDX   29884416u   // NL_ int
#define WS_ACC   29949952u   // 4 f32 loss accumulators

__device__ inline u16 bf16_rne(float f) {
  u32 u = __float_as_uint(f);
  u32 r = u + 0x7fffu + ((u >> 16) & 1u);
  return (u16)(r >> 16);
}

union FragB16 { v16bf v; u32x4 q[2]; u16 h[16]; };

// load 4 B fragments (8 x global_load_b128) starting at k-step s0
__device__ inline void load_half(FragB16* dst, const u16* brow, int s0) {
#pragma unroll
  for (int s = 0; s < 4; ++s) {
    const u16* p = brow + (s0 + s) * 32;
    dst[s].q[0] = ((const u32x4*)p)[0];
    dst[s].q[1] = ((const u32x4*)p)[1];
  }
}

// --------------------------------------------------------------------------
// Fused sim GEMM + argmax: block = 16 positions, 8 waves sweep K=8192 codes.
// A frags (16x256 bf16) live in VGPRs; B double-buffered per half-pass.
__global__ void __launch_bounds__(256) vq_sim(
    const u16* __restrict__ zen, const u16* __restrict__ cbn,
    int* __restrict__ idx_ws, float* __restrict__ out_idx) {
  const int t = threadIdx.x, lane = t & 31, wave = t >> 5;
  const int pb = blockIdx.x * 16;
  const int hi = (lane >= 16) ? 1 : 0;
  const int mcol = lane & 15;

  FragB16 a[8];
  const u16* zr = zen + (size_t)(pb + mcol) * D_;
#pragma unroll
  for (int s = 0; s < 8; ++s) {
    const u16* ap = zr + s * 32 + hi * 16;
    a[s].q[0] = *(const u32x4*)ap;
    a[s].q[1] = *(const u32x4*)(ap + 8);
  }

  float bestv[8]; int besti[8];
#pragma unroll
  for (int v = 0; v < 8; ++v) { bestv[v] = -3.0e38f; besti[v] = 0; }

  // per-pass row pointer: cbn[(pass*128 + wave*16 + mcol) * D_ + hi*16]
  const u16* base = cbn + (size_t)(wave * 16 + mcol) * D_ + hi * 16;

  FragB16 b0[4], b1[4];
  load_half(b0, base, 0);                        // prologue: pass 0, first half

  for (int pass = 0; pass < 64; ++pass) {
    const u16* cur = base + (size_t)pass * 128 * D_;
    const u16* nxt = cur + 128 * D_;             // pass 63 overreads into zews: benign
    __builtin_prefetch(nxt + 128 * D_, 0, 1);    // global_prefetch two passes ahead

    load_half(b1, cur, 4);                       // 2nd half of current pass

    v8f acc;
#pragma unroll
    for (int v = 0; v < 8; ++v) acc[v] = 0.f;
#pragma unroll
    for (int s = 0; s < 4; ++s)
      acc = __builtin_amdgcn_wmma_f32_16x16x32_bf16(
          false, a[s].v, false, b0[s].v, (short)0, acc, false, false);

    load_half(b0, nxt, 0);                       // 1st half of next pass

#pragma unroll
    for (int s = 0; s < 4; ++s)
      acc = __builtin_amdgcn_wmma_f32_16x16x32_bf16(
          false, a[4 + s].v, false, b1[s].v, (short)0, acc, false, false);

    const int code = pass * 128 + wave * 16 + mcol;  // monotonic -> first-max
#pragma unroll
    for (int v = 0; v < 8; ++v)
      if (acc[v] > bestv[v]) { bestv[v] = acc[v]; besti[v] = code; }
  }

  __shared__ float s_val[8][16][16];
  __shared__ int   s_idx[8][16][16];
#pragma unroll
  for (int v = 0; v < 8; ++v) {
    const int m = v + hi * 8;
    s_val[wave][m][mcol] = bestv[v];
    s_idx[wave][m][mcol] = besti[v];
  }
  __syncthreads();
  if (t < 16) {
    float bv = -3.0e38f; int bi = 0x7fffffff;
    for (int w = 0; w < 8; ++w)
      for (int c = 0; c < 16; ++c) {
        const float v = s_val[w][t][c]; const int i = s_idx[w][t][c];
        if (v > bv || (v == bv && i < bi)) { bv = v; bi = i; }
      }
    idx_ws[pb + t]  = bi;
    out_idx[pb + t] = (float)bi;
  }
}

// --------------------------------------------------------------------------
// z_e[p, d] = sum_c x[n, c, l] * W_in[d, c] + in_b[d]   (p = n*L + l)
__global__ void __launch_bounds__(128) vq_ze(
    const float* __restrict__ x, const u16* __restrict__ w_in_bf,
    const float* __restrict__ in_b, float* __restrict__ zews) {
  const int t = threadIdx.x;
  const int lane = t & 31, wave = t >> 5;
  const int pb = blockIdx.x * 16;
  const int nbatch = pb >> 12;            // / L_
  const int l0 = pb & (L_ - 1);
  const int hi = (lane >= 16) ? 1 : 0;
  const int mcol = lane & 15;

  __shared__ __align__(16) u16 xt[16][32];

  v8f acc[4];
#pragma unroll
  for (int f = 0; f < 4; ++f)
#pragma unroll
    for (int v = 0; v < 8; ++v) acc[f][v] = 0.f;

  const int q  = t & 3;                   // which quad of rows this lane stages
  const int cc = t >> 2;                  // 0..31: channel within tile
  const float* xg = x + (size_t)nbatch * C_ * L_ + l0 + q * 4;

  for (int kk = 0; kk < C_; kk += 32) {
    __syncthreads();
    f32x4 xv = *(const f32x4*)(xg + (size_t)(kk + cc) * L_);
    xt[q * 4 + 0][cc] = bf16_rne(xv.x);
    xt[q * 4 + 1][cc] = bf16_rne(xv.y);
    xt[q * 4 + 2][cc] = bf16_rne(xv.z);
    xt[q * 4 + 3][cc] = bf16_rne(xv.w);
    __syncthreads();

    FragB16 b[4];
#pragma unroll
    for (int f = 0; f < 4; ++f) {
      const int ncol = wave * 64 + f * 16 + mcol;      // output channel d
      const u16* bp = w_in_bf + (size_t)ncol * C_ + kk + hi * 16;
      b[f].q[0] = ((const u32x4*)bp)[0];
      b[f].q[1] = ((const u32x4*)bp)[1];
    }
    FragB16 a;                            // 16x32 A frag (contiguous k = 16*hi + s)
    a.q[0] = *(const u32x4*)&xt[mcol][hi * 16];
    a.q[1] = *(const u32x4*)&xt[mcol][hi * 16 + 8];

#pragma unroll
    for (int f = 0; f < 4; ++f)
      acc[f] = __builtin_amdgcn_wmma_f32_16x16x32_bf16(
          false, a.v, false, b[f].v, (short)0, acc[f], false, false);
  }
#pragma unroll
  for (int f = 0; f < 4; ++f) {
    const int dcol = wave * 64 + f * 16 + mcol;
    const float bias = in_b[dcol];
#pragma unroll
    for (int v = 0; v < 8; ++v) {
      const int p = pb + v + hi * 8;
      zews[(size_t)p * D_ + dcol] = acc[f][v] + bias;
    }
  }
}

// --------------------------------------------------------------------------
// y[n, c, l] = sum_d codebook[idx[p], d] * W_out[c, d] + out_b[c]
__global__ void __launch_bounds__(256) vq_y(
    const float* __restrict__ codebook, const u16* __restrict__ w_out_bf,
    const float* __restrict__ out_b, const int* __restrict__ idx_ws,
    float* __restrict__ y) {
  const int t = threadIdx.x, lane = t & 31, wave = t >> 5;
  const int pb = blockIdx.x * 16;
  const int hi = (lane >= 16) ? 1 : 0;
  const int mcol = lane & 15;

  __shared__ int s_code[16];
  __shared__ __align__(16) u16 zq[16][256];
  if (t < 16) s_code[t] = idx_ws[pb + t];
  __syncthreads();
  {
    const int r = t & 15, cb0 = (t >> 4) * 16;
    const float* src = codebook + (size_t)s_code[r] * D_ + cb0;
    union { u32x4 q[2]; u16 h[16]; } pk;
#pragma unroll
    for (int j = 0; j < 16; j += 4) {
      f32x4 f = *(const f32x4*)(src + j);
      pk.h[j+0] = bf16_rne(f.x); pk.h[j+1] = bf16_rne(f.y);
      pk.h[j+2] = bf16_rne(f.z); pk.h[j+3] = bf16_rne(f.w);
    }
    *(u32x4*)&zq[r][cb0]     = pk.q[0];
    *(u32x4*)&zq[r][cb0 + 8] = pk.q[1];
  }
  __syncthreads();

  v8f acc[4];
#pragma unroll
  for (int f = 0; f < 4; ++f)
#pragma unroll
    for (int v = 0; v < 8; ++v) acc[f][v] = 0.f;

  for (int kk = 0; kk < D_; kk += 32) {
    FragB16 b[4];
#pragma unroll
    for (int f = 0; f < 4; ++f) {
      const int ccol = wave * 64 + f * 16 + mcol;       // output channel c
      const u16* bp = w_out_bf + (size_t)ccol * D_ + kk + hi * 16;
      b[f].q[0] = ((const u32x4*)bp)[0];
      b[f].q[1] = ((const u32x4*)bp)[1];
    }
    FragB16 a;
    a.q[0] = *(const u32x4*)&zq[mcol][kk + hi * 16];
    a.q[1] = *(const u32x4*)&zq[mcol][kk + hi * 16 + 8];
#pragma unroll
    for (int f = 0; f < 4; ++f)
      acc[f] = __builtin_amdgcn_wmma_f32_16x16x32_bf16(
          false, a.v, false, b[f].v, (short)0, acc[f], false, false);
  }
  const int nbatch = pb >> 12, l0 = pb & (L_ - 1);
#pragma unroll
  for (int f = 0; f < 4; ++f) {
    const int ccol = wave * 64 + f * 16 + mcol;
    const float bias = out_b[ccol];
    float* yrow = y + (size_t)nbatch * C_ * L_ + (size_t)ccol * L_ + l0;
#pragma unroll
    for (int v = 0; v < 8; ++v)
      yrow[v + hi * 8] = acc[f][v] + bias;
  }
}

// --------------------------------------------------------------------------
// per-position L2 normalize z_e rows -> bf16 (one wave per position)
__global__ void __launch_bounds__(256) vq_norm(
    const float* __restrict__ zews, u16* __restrict__ zen) {
  const int lane = threadIdx.x & 31, wave = threadIdx.x >> 5;
  const int p = blockIdx.x * 8 + wave;
  const float* zr = zews + (size_t)p * D_ + lane * 8;
  f32x4 a = ((const f32x4*)zr)[0];
  f32x4 b = ((const f32x4*)zr)[1];
  float ss = a.x*a.x + a.y*a.y + a.z*a.z + a.w*a.w
           + b.x*b.x + b.y*b.y + b.z*b.z + b.w*b.w;
  for (int o = 16; o > 0; o >>= 1) ss += __shfl_xor(ss, o, 32);
  const float inv = 1.f / fmaxf(sqrtf(ss), EPS_);
  union { u32x4 q; u16 h[8]; } pk;
  pk.h[0] = bf16_rne(a.x*inv); pk.h[1] = bf16_rne(a.y*inv);
  pk.h[2] = bf16_rne(a.z*inv); pk.h[3] = bf16_rne(a.w*inv);
  pk.h[4] = bf16_rne(b.x*inv); pk.h[5] = bf16_rne(b.y*inv);
  pk.h[6] = bf16_rne(b.z*inv); pk.h[7] = bf16_rne(b.w*inv);
  *(u32x4*)(zen + (size_t)p * D_ + lane * 8) = pk.q;
}

// --------------------------------------------------------------------------
__device__ inline float block_sum_256(float v) {
  __shared__ float sh[8];
  for (int o = 16; o > 0; o >>= 1) v += __shfl_down(v, o, 32);
  int lane = threadIdx.x & 31, w = threadIdx.x >> 5;
  if (lane == 0) sh[w] = v;
  __syncthreads();
  float r = (threadIdx.x < 8) ? sh[threadIdx.x] : 0.f;
  if (w == 0) for (int o = 4; o > 0; o >>= 1) r += __shfl_down(r, o, 32);
  __shared__ float tot;
  if (threadIdx.x == 0) tot = r;
  __syncthreads();
  return tot;
}

// one block per row: weight-norm both convs + normalize codebook, emit bf16
__global__ void __launch_bounds__(256) vq_prep(
    const float* __restrict__ in_v,  const float* __restrict__ in_g,
    const float* __restrict__ out_v, const float* __restrict__ out_g,
    const float* __restrict__ codebook,
    u16* __restrict__ w_in_bf, u16* __restrict__ w_out_bf,
    u16* __restrict__ cbn_bf) {
  const int r = blockIdx.x, t = threadIdx.x;
  if (r < D_) {                               // W_in row: length C_=512
    float a = in_v[(size_t)r * C_ + t];
    float b = in_v[(size_t)r * C_ + t + 256];
    float ss = block_sum_256(a * a + b * b);
    float scale = in_g[r] * rsqrtf(ss);       // g * v / ||v||
    w_in_bf[(size_t)r * C_ + t]       = bf16_rne(a * scale);
    w_in_bf[(size_t)r * C_ + t + 256] = bf16_rne(b * scale);
  } else if (r < D_ + C_) {                   // W_out row: length D_=256
    int rr = r - D_;
    float a = out_v[(size_t)rr * D_ + t];
    float ss = block_sum_256(a * a);
    float scale = out_g[rr] * rsqrtf(ss);
    w_out_bf[(size_t)rr * D_ + t] = bf16_rne(a * scale);
  } else {                                    // codebook row normalize
    int rr = r - D_ - C_;
    float a = codebook[(size_t)rr * D_ + t];
    float ss = block_sum_256(a * a);
    float scale = 1.f / fmaxf(sqrtf(ss), EPS_);
    cbn_bf[(size_t)rr * D_ + t] = bf16_rne(a * scale);
  }
}

// --------------------------------------------------------------------------
__global__ void vq_init(float* acc4) {
  if (threadIdx.x < 4) acc4[threadIdx.x] = 0.f;
}

__global__ void __launch_bounds__(256) vq_loss(
    const float* __restrict__ codebook, const float* __restrict__ zews,
    const int* __restrict__ idx_ws, float* __restrict__ acc4) {
  const int p = blockIdx.x * 256 + threadIdx.x;
  const float* cb = codebook + (size_t)idx_ws[p] * D_;
  const float* ze = zews + (size_t)p * D_;
  float s = 0.f;
  for (int d = 0; d < D_; d += 4) {
    f32x4 a = *(const f32x4*)(cb + d);
    f32x4 b = *(const f32x4*)(ze + d);
    float dx = a.x-b.x, dy = a.y-b.y, dz = a.z-b.z, dw = a.w-b.w;
    s += dx*dx + dy*dy + dz*dz + dw*dw;
  }
  atomicAdd(&acc4[p >> 12], s);
}

__global__ void vq_fin(const float* __restrict__ acc4, float* __restrict__ out_loss) {
  const int t = threadIdx.x;
  if (t < 4) {
    const float v = acc4[t] / (float)(D_ * L_);
    out_loss[t]     = v;   // cb_loss
    out_loss[4 + t] = v;   // cm_loss (numerically identical)
  }
}

// --------------------------------------------------------------------------
extern "C" void kernel_launch(void* const* d_in, const int* in_sizes, int n_in,
                              void* d_out, int out_size, void* d_ws, size_t ws_size,
                              hipStream_t stream) {
  const float* x     = (const float*)d_in[0];
  const float* in_v  = (const float*)d_in[1];
  const float* in_g  = (const float*)d_in[2];
  const float* in_b  = (const float*)d_in[3];
  const float* out_v = (const float*)d_in[4];
  const float* out_g = (const float*)d_in[5];
  const float* out_b = (const float*)d_in[6];
  const float* cbk   = (const float*)d_in[7];
  float* out = (float*)d_out;

  char* ws = (char*)d_ws;
  u16*   w_in_bf  = (u16*)(ws + WS_WIN);
  u16*   w_out_bf = (u16*)(ws + WS_WOUT);
  u16*   cbn_bf   = (u16*)(ws + WS_CBN);
  float* zews     = (float*)(ws + WS_ZE);
  u16*   zen      = (u16*)(ws + WS_ZEN);
  int*   idx_ws   = (int*)(ws + WS_IDX);
  float* acc4     = (float*)(ws + WS_ACC);

  vq_init<<<1, 32, 0, stream>>>(acc4);
  vq_prep<<<D_ + C_ + K_, 256, 0, stream>>>(in_v, in_g, out_v, out_g, cbk,
                                            w_in_bf, w_out_bf, cbn_bf);
  vq_ze  <<<NL_ / 16, 128, 0, stream>>>(x, w_in_bf, in_b, zews);
  vq_norm<<<NL_ / 8, 256, 0, stream>>>(zews, zen);
  vq_sim <<<NL_ / 16, 256, 0, stream>>>(zen, cbn_bf, idx_ws, out + Y_SIZE_);
  vq_y   <<<NL_ / 16, 256, 0, stream>>>(cbk, w_out_bf, out_b, idx_ws, out);
  vq_loss<<<NL_ / 256, 256, 0, stream>>>(cbk, zews, idx_ws, acc4);
  vq_fin <<<1, 32, 0, stream>>>(acc4, out + Y_SIZE_ + NL_);
}
